// SMRU_35356170780818
// MI455X (gfx1250) — matrix operand order repaired and checked by reading, not compiled, not run on hardware
//
#include <hip/hip_runtime.h>

#define T_STEPS 2048
#define BATCH   32
#define HDIM    1024
#define G3H     (3 * HDIM)
#define COLS    16                 // features per workgroup (recurrent phase)
#define KTOT    2048               // fused-path combined K
#define NWG     (HDIM / COLS)      // 64 workgroups
#define NTHR    64                 // 2 waves (wave32)
#define GEMM_THR  128              // 4 waves
#define GEMM_MSPL 64               // M splits in phase-1 grid

typedef __attribute__((ext_vector_type(16))) __bf16 v16bf;
typedef __attribute__((ext_vector_type(8)))  float  v8f;
typedef __attribute__((ext_vector_type(4)))  unsigned v4u;
typedef __attribute__((ext_vector_type(8)))  int      v8i;
typedef __attribute__((ext_vector_type(4)))  int      v4i;

#if defined(__has_builtin)
#  if __has_builtin(__builtin_amdgcn_tensor_load_to_lds) && \
      __has_builtin(__builtin_amdgcn_s_wait_tensorcnt)
#    define USE_TDM 1
#  endif
#endif
#ifndef USE_TDM
#  define USE_TDM 0
#endif

__device__ __forceinline__ unsigned short f2bf(float f) {
  unsigned u = __builtin_bit_cast(unsigned, f);
  u += 0x7FFFu + ((u >> 16) & 1u);               // round-to-nearest-even
  return (unsigned short)(u >> 16);
}
__device__ __forceinline__ float bf2f(unsigned short s) {
  unsigned u = ((unsigned)s) << 16;
  return __builtin_bit_cast(float, u);
}

union Frag {
  uint4          q[2];
  unsigned short s[16];
  v16bf          v;
};

// A-matrix bf16 fragment: lane half holds K {c0..c0+15}, contiguous,
// caller folds c0 = 16*(lane>=16) into p.
__device__ __forceinline__ v16bf ld_fragA_b16(const unsigned short* p) {
  Frag u;
  u.q[0] = *(const uint4*)(p);
  u.q[1] = *(const uint4*)(p + 8);
  return u.v;
}

// B-matrix bf16 fragment (this code's order, same as A): lanes 0-15 hold K 0-15, lanes 16-31 hold
// K 16-31, contiguous; caller folds 16*(lane>=16) into p.
__device__ __forceinline__ v16bf ld_fragB_b16(const unsigned short* p) {
  Frag u;
  u.q[0] = *(const uint4*)(p);
  u.q[1] = *(const uint4*)(p + 8);
  return u.v;
}

// f32 A-fragment with on-the-fly bf16 conversion (phase-1 / fused only)
__device__ __forceinline__ v16bf cvt_fragA_f32(const float* p) {
  Frag u;
  const float4 f0 = *(const float4*)(p + 0);
  const float4 f1 = *(const float4*)(p + 4);
  const float4 f2 = *(const float4*)(p + 8);
  const float4 f3 = *(const float4*)(p + 12);
  u.s[0]  = f2bf(f0.x); u.s[1]  = f2bf(f0.y); u.s[2]  = f2bf(f0.z); u.s[3]  = f2bf(f0.w);
  u.s[4]  = f2bf(f1.x); u.s[5]  = f2bf(f1.y); u.s[6]  = f2bf(f1.z); u.s[7]  = f2bf(f1.w);
  u.s[8]  = f2bf(f2.x); u.s[9]  = f2bf(f2.y); u.s[10] = f2bf(f2.z); u.s[11] = f2bf(f2.w);
  u.s[12] = f2bf(f3.x); u.s[13] = f2bf(f3.y); u.s[14] = f2bf(f3.z); u.s[15] = f2bf(f3.w);
  return u.v;
}

__device__ __forceinline__ v8f wmma_bf16(v16bf a, v16bf b, v8f c) {
  return __builtin_amdgcn_wmma_f32_16x16x32_bf16(false, a, false, b, (short)0, c,
                                                 false, false);
}

__device__ __forceinline__ void grid_sync(unsigned* cnt, volatile unsigned* gen) {
  __threadfence();
  __syncthreads();
  if (threadIdx.x == 0) {
    const unsigned g = *gen;
    const unsigned arrived = atomicAdd(cnt, 1u) + 1u;
    if (arrived == (unsigned)NWG) {
      *cnt = 0u;
      __threadfence();
      *gen = g + 1u;
    } else {
      while (*gen == g) { __builtin_amdgcn_s_sleep(2); }
    }
  }
  __syncthreads();
  __threadfence();
}

#if USE_TDM
// Tensor Data Mover: DMA the full bf16 h state (32x1024 = 64KB, contiguous) into
// LDS offset 0 (hstage is first in the dynamic-LDS block). D# per ISA 8.3/8.4.
__device__ __forceinline__ void tdm_load_h(const unsigned short* gsrc) {
  const unsigned long long ga = (unsigned long long)(size_t)gsrc;
  v4u g0;
  g0.x = 0x1u;                                           // count=1 (valid user D#)
  g0.y = 0u;                                             // lds_addr = 0
  g0.z = (unsigned)(ga & 0xFFFFFFFFull);                 // global_addr[31:0]
  g0.w = (unsigned)((ga >> 32) & 0x01FFFFFFull) | 0x80000000u;  // [56:32] | type=2
  v8i g1;
  g1[0] = 0x00010000;          // wg_mask=0, data_size=1 (2 bytes)
  g1[1] = (int)0x80000000;     // tensor_dim0 = 32768 (bits 79:48, low16 here)
  g1[2] = 0x00010000;          // tensor_dim0[31:16]=0, tensor_dim1 = 1
  g1[3] = (int)0x80000000;     // tile_dim0 = 32768 (bits 127:112)
  g1[4] = 0;                   // tile_dim1 = 0 (unused), tile_dim2 = 0
  g1[5] = 32768;               // tensor_dim0_stride[31:0]
  g1[6] = 0;
  g1[7] = 0;
  v4i z4 = {0, 0, 0, 0};
#if __clang_major__ >= 23
  v8i z8 = {0, 0, 0, 0, 0, 0, 0, 0};
  __builtin_amdgcn_tensor_load_to_lds(g0, g1, z4, z4, z8, 0);
#else
  __builtin_amdgcn_tensor_load_to_lds(g0, g1, z4, z4, 0);
#endif
}
#endif

__global__ void smru_init_kernel(const float* __restrict__ h0,
                                 unsigned short* __restrict__ hbufA,
                                 unsigned* __restrict__ sync) {
  const int i = blockIdx.x * blockDim.x + threadIdx.x;
  if (i < BATCH * HDIM) hbufA[i] = f2bf(h0[i]);
  if (i < 16) sync[i] = 0u;
}

// ---------------- Phase 1: G[m, n] = x[m, :] . w_ih[n, :] ----------------
__global__ void __launch_bounds__(GEMM_THR, 1)
smru_gemm_kernel(const float* __restrict__ x,
                 const float* __restrict__ w_ih,
                 float* __restrict__ G) {
  extern __shared__ unsigned short Bt[];   // [64 rows][1024 K] bf16 = 128 KB
  const int tid = threadIdx.x;
  const int n0  = blockIdx.x * 64;

  for (int gidx = tid; gidx < 64 * HDIM / 4; gidx += GEMM_THR) {
    const int idx = gidx * 4;
    const int row = idx / HDIM;
    const int k   = idx - row * HDIM;
    const float4 v = *(const float4*)(w_ih + (size_t)(n0 + row) * HDIM + k);
    Bt[idx + 0] = f2bf(v.x);
    Bt[idx + 1] = f2bf(v.y);
    Bt[idx + 2] = f2bf(v.z);
    Bt[idx + 3] = f2bf(v.w);
  }
  __syncthreads();

  const int lane  = tid & 31;
  const int l15   = lane & 15;
  const int khalf = lane >> 4;
  const int wv    = tid >> 5;

  const unsigned short* B0 = Bt + (0 * 16 + l15) * HDIM + khalf * 16;
  const unsigned short* B1 = Bt + (1 * 16 + l15) * HDIM + khalf * 16;
  const unsigned short* B2 = Bt + (2 * 16 + l15) * HDIM + khalf * 16;
  const unsigned short* B3 = Bt + (3 * 16 + l15) * HDIM + khalf * 16;

  const int iters = (T_STEPS * BATCH) / (64 * GEMM_MSPL);   // 16
  for (int it = 0; it < iters; ++it) {
    const int m0 = (blockIdx.y * iters + it) * 64 + wv * 16;
    v8f a0 = {0.f,0.f,0.f,0.f,0.f,0.f,0.f,0.f};
    v8f a1 = a0, a2 = a0, a3 = a0;
    const float* xr = x + (size_t)(m0 + l15) * HDIM + khalf * 16;
#pragma unroll 2
    for (int kk = 0; kk < 32; ++kk) {
      const v16bf a  = cvt_fragA_f32(xr + kk * 32);
      const v16bf b0 = ld_fragB_b16(B0 + kk * 32);
      const v16bf b1 = ld_fragB_b16(B1 + kk * 32);
      const v16bf b2 = ld_fragB_b16(B2 + kk * 32);
      const v16bf b3 = ld_fragB_b16(B3 + kk * 32);
      a0 = wmma_bf16(a, b0, a0);
      a1 = wmma_bf16(a, b1, a1);
      a2 = wmma_bf16(a, b2, a2);
      a3 = wmma_bf16(a, b3, a3);
    }
    float* gp = G + (size_t)(m0 + khalf * 8) * G3H + n0 + l15;
#pragma unroll
    for (int e = 0; e < 8; ++e) {
      float* r = gp + (size_t)e * G3H;
      r[0] = a0[e]; r[16] = a1[e]; r[32] = a2[e]; r[48] = a3[e];
    }
  }
}

// ---------------- Phase 2: serial recurrence, h-GEMM only ----------------
__global__ void __launch_bounds__(NTHR, 1)
smru_rec_kernel(const float* __restrict__ G,
                const float* __restrict__ w_hh,
                const float* __restrict__ b_ih,
                const float* __restrict__ b_hh,
                float* __restrict__ out,
                unsigned short* __restrict__ hbufA,
                unsigned short* __restrict__ hbufB,
                unsigned* __restrict__ sync) {
  extern __shared__ unsigned short smem[];
  unsigned short* hstage = smem;                    // [32][1024] bf16 @ LDS 0
  unsigned short* Wh     = smem + BATCH * HDIM;     // [3][COLS][1024] bf16

  const int tid = threadIdx.x;
  const int j0  = blockIdx.x * COLS;

  // one-time: W_hh slice -> bf16 LDS
  for (int gidx = tid; gidx < 3 * COLS * HDIM / 4; gidx += NTHR) {
    const int idx = gidx * 4;
    const int g   = idx / (COLS * HDIM);
    const int r   = idx - g * (COLS * HDIM);
    const int col = r / HDIM;
    const int k   = r - col * HDIM;
    const float4 v = *(const float4*)(w_hh + (size_t)(g * HDIM + j0 + col) * HDIM + k);
    Wh[idx + 0] = f2bf(v.x);
    Wh[idx + 1] = f2bf(v.y);
    Wh[idx + 2] = f2bf(v.z);
    Wh[idx + 3] = f2bf(v.w);
  }
  __syncthreads();

  const int lane  = tid & 31;
  const int l15   = lane & 15;
  const int khalf = lane >> 4;
  const int wv    = tid >> 5;
  const int b0    = wv * 16;
  const int j     = j0 + l15;

  const float bk  = b_ih[j]            + b_hh[j];
  const float bfv = b_ih[j + HDIM]     + b_hh[j + HDIM];
  const float ba  = b_ih[j + 2 * HDIM] + b_hh[j + 2 * HDIM];

  const unsigned short* Wk = Wh + 0 * COLS * HDIM + l15 * HDIM + khalf * 16;
  const unsigned short* Wf = Wh + 1 * COLS * HDIM + l15 * HDIM + khalf * 16;
  const unsigned short* Wa = Wh + 2 * COLS * HDIM + l15 * HDIM + khalf * 16;

  unsigned short* hb[2] = { hbufA, hbufB };
  int p = 0;

  for (int t = 0; t < T_STEPS; ++t) {
    // ---- stage h[t] into LDS ----
#if USE_TDM
    if (tid < 32) {
      tdm_load_h(hb[p]);
      __builtin_amdgcn_s_wait_tensorcnt((short)0);
    }
#else
    for (int i = tid; i < BATCH * HDIM / 8; i += NTHR)
      ((uint4*)hstage)[i] = ((const uint4*)hb[p])[i];
#endif
    __syncthreads();

    const float* Gp = G + (size_t)t * BATCH * G3H;
    if (t + 1 < T_STEPS)
      __builtin_prefetch(Gp + (size_t)BATCH * G3H + j, 0, 1);

    // accumulators initialized from precomputed x-projection (C layout match)
    v8f ak, af, aa;
#pragma unroll
    for (int e = 0; e < 8; ++e) {
      const float* gr = Gp + (size_t)(b0 + khalf * 8 + e) * G3H + j;
      ak[e] = gr[0];
      af[e] = gr[HDIM];
      aa[e] = gr[2 * HDIM];
    }

    // ---- h @ W_hh^T, software-pipelined A fragment ----
    const unsigned short* hr = hstage + (b0 + l15) * HDIM + khalf * 16;
    v16bf acur = ld_fragA_b16(hr);
#pragma unroll 4
    for (int kk = 0; kk < 32; ++kk) {
      const v16bf bk_ = ld_fragB_b16(Wk + kk * 32);
      const v16bf bf_ = ld_fragB_b16(Wf + kk * 32);
      const v16bf ba_ = ld_fragB_b16(Wa + kk * 32);
      const v16bf anx = ld_fragA_b16(hr + ((kk + 1) & 31) * 32);
      ak = wmma_bf16(acur, bk_, ak);
      af = wmma_bf16(acur, bf_, af);
      aa = wmma_bf16(acur, ba_, aa);
      acur = anx;
    }

    // ---- 3-way softmax epilogue, h' = k_sm * h + a_sm ----
    unsigned short* hw = hb[p ^ 1];
    float* op = out + (size_t)t * BATCH * HDIM;
#pragma unroll
    for (int e = 0; e < 8; ++e) {
      const int br = b0 + khalf * 8 + e;
      const float kv = ak[e] + bk;
      const float fv = af[e] + bfv;
      const float av = aa[e] + ba;
      const float mx = fmaxf(kv, fmaxf(fv, av));
      const float ek = __expf(kv - mx);
      const float ef = __expf(fv - mx);
      const float ea = __expf(av - mx);
      const float inv = 1.0f / (ek + ef + ea);
      const float hp  = bf2f(hstage[br * HDIM + j]);
      const float hy  = (ek * hp + ea) * inv;
      op[(size_t)br * HDIM + j] = hy;
      hw[(size_t)br * HDIM + j] = f2bf(hy);
    }

    grid_sync(sync, sync + 1);
    p ^= 1;
  }
}

// ---------------- Fallback: fused single-pass kernel (small workspace) ----------------
__global__ void __launch_bounds__(NTHR, 1)
smru_fused_kernel(const float* __restrict__ x,
                  const float* __restrict__ w_ih,
                  const float* __restrict__ w_hh,
                  const float* __restrict__ b_ih,
                  const float* __restrict__ b_hh,
                  float* __restrict__ out,
                  unsigned short* __restrict__ hbufA,
                  unsigned short* __restrict__ hbufB,
                  unsigned* __restrict__ sync) {
  extern __shared__ unsigned short Wlds[];   // [3][COLS][KTOT] bf16 = 192 KB
  const int tid = threadIdx.x;
  const int j0  = blockIdx.x * COLS;

  const int groups = 3 * COLS * KTOT / 4;
  for (int gidx = tid; gidx < groups; gidx += NTHR) {
    const int idx = gidx * 4;
    const int g   = idx / (COLS * KTOT);
    const int r   = idx - g * (COLS * KTOT);
    const int col = r / KTOT;
    const int k   = r - col * KTOT;
    const int row = g * HDIM + j0 + col;
    float4 v;
    if (k < HDIM) v = *(const float4*)(w_ih + (size_t)row * HDIM + k);
    else          v = *(const float4*)(w_hh + (size_t)row * HDIM + (k - HDIM));
    Wlds[idx + 0] = f2bf(v.x);
    Wlds[idx + 1] = f2bf(v.y);
    Wlds[idx + 2] = f2bf(v.z);
    Wlds[idx + 3] = f2bf(v.w);
  }
  __syncthreads();

  const int lane  = tid & 31;
  const int l15   = lane & 15;
  const int khalf = lane >> 4;
  const int wv    = tid >> 5;
  const int b0    = wv * 16;
  const int j     = j0 + l15;

  const float bk  = b_ih[j]            + b_hh[j];
  const float bfv = b_ih[j + HDIM]     + b_hh[j + HDIM];
  const float ba  = b_ih[j + 2 * HDIM] + b_hh[j + 2 * HDIM];

  const unsigned short* Wk = Wlds + 0 * COLS * KTOT + l15 * KTOT + khalf * 16;
  const unsigned short* Wf = Wlds + 1 * COLS * KTOT + l15 * KTOT + khalf * 16;
  const unsigned short* Wa = Wlds + 2 * COLS * KTOT + l15 * KTOT + khalf * 16;

  unsigned short* hb[2] = { hbufA, hbufB };
  int p = 0;

  for (int t = 0; t < T_STEPS; ++t) {
    v8f ak = {0.f,0.f,0.f,0.f,0.f,0.f,0.f,0.f};
    v8f af = ak, aa = ak;

    const float* xr = x + ((size_t)t * BATCH + b0 + l15) * HDIM + khalf * 16;
    if (t + 1 < T_STEPS)
      __builtin_prefetch(x + ((size_t)(t + 1) * BATCH + b0 + l15) * HDIM, 0, 1);
#pragma unroll 4
    for (int kk = 0; kk < 32; ++kk) {
      const v16bf a   = cvt_fragA_f32(xr + kk * 32);
      const v16bf bk_ = ld_fragB_b16(Wk + kk * 32);
      const v16bf bf_ = ld_fragB_b16(Wf + kk * 32);
      const v16bf ba_ = ld_fragB_b16(Wa + kk * 32);
      ak = wmma_bf16(a, bk_, ak);
      af = wmma_bf16(a, bf_, af);
      aa = wmma_bf16(a, ba_, aa);
    }
    const unsigned short* hr = hb[p] + (size_t)(b0 + l15) * HDIM + khalf * 16;
#pragma unroll 4
    for (int kk = 0; kk < 32; ++kk) {
      const v16bf a   = ld_fragA_b16(hr + kk * 32);
      const v16bf bk_ = ld_fragB_b16(Wk + HDIM + kk * 32);
      const v16bf bf_ = ld_fragB_b16(Wf + HDIM + kk * 32);
      const v16bf ba_ = ld_fragB_b16(Wa + HDIM + kk * 32);
      ak = wmma_bf16(a, bk_, ak);
      af = wmma_bf16(a, bf_, af);
      aa = wmma_bf16(a, ba_, aa);
    }

    const unsigned short* hc = hb[p];
    unsigned short*       hw = hb[p ^ 1];
    float* op = out + (size_t)t * BATCH * HDIM;
#pragma unroll
    for (int e = 0; e < 8; ++e) {
      const int br = b0 + khalf * 8 + e;
      const float kv = ak[e] + bk;
      const float fv = af[e] + bfv;
      const float av = aa[e] + ba;
      const float mx = fmaxf(kv, fmaxf(fv, av));
      const float ek = __expf(kv - mx);
      const float ef = __expf(fv - mx);
      const float ea = __expf(av - mx);
      const float inv = 1.0f / (ek + ef + ea);
      const float hp  = bf2f(hc[(size_t)br * HDIM + j]);
      const float hy  = (ek * hp + ea) * inv;
      op[(size_t)br * HDIM + j] = hy;
      hw[(size_t)br * HDIM + j] = f2bf(hy);
    }

    grid_sync(sync, sync + 1);
    p ^= 1;
  }
}

extern "C" void kernel_launch(void* const* d_in, const int* in_sizes, int n_in,
                              void* d_out, int out_size, void* d_ws, size_t ws_size,
                              hipStream_t stream) {
  const float* x    = (const float*)d_in[0];
  const float* h0   = (const float*)d_in[1];
  const float* w_ih = (const float*)d_in[2];
  const float* w_hh = (const float*)d_in[3];
  const float* b_ih = (const float*)d_in[4];
  const float* b_hh = (const float*)d_in[5];
  float* out = (float*)d_out;

  unsigned*       sync = (unsigned*)d_ws;
  unsigned short* hA   = (unsigned short*)((char*)d_ws + 256);
  unsigned short* hB   = hA + (size_t)BATCH * HDIM;

  size_t goff = 256 + 2 * (size_t)BATCH * HDIM * sizeof(unsigned short);
  goff = (goff + 511) & ~(size_t)511;
  const size_t gbytes = (size_t)T_STEPS * BATCH * G3H * sizeof(float);  // 768 MB

  smru_init_kernel<<<(BATCH * HDIM + 255) / 256, 256, 0, stream>>>(h0, hA, sync);

  if (ws_size >= goff + gbytes) {
    float* G = (float*)((char*)d_ws + goff);
    dim3 ggrid(G3H / 64, GEMM_MSPL);
    smru_gemm_kernel<<<ggrid, GEMM_THR, (size_t)64 * HDIM * 2, stream>>>(x, w_ih, G);
    const size_t rec_lds = ((size_t)BATCH * HDIM + 3 * COLS * HDIM) * 2;  // 160 KB
    smru_rec_kernel<<<NWG, NTHR, rec_lds, stream>>>(G, w_hh, b_ih, b_hh,
                                                    out, hA, hB, sync);
  } else {
    const size_t lds_bytes = (size_t)3 * COLS * KTOT * 2;                 // 192 KB
    smru_fused_kernel<<<NWG, NTHR, lds_bytes, stream>>>(x, w_ih, w_hh, b_ih, b_hh,
                                                        out, hA, hB, sync);
  }
}
